// Seq2Seq_52407190945916
// MI455X (gfx1250) — matrix operand order repaired and checked by reading, not compiled, not run on hardware
//
#include <hip/hip_runtime.h>
#include <hip/hip_bf16.h>
#include <stdint.h>

// Model dims (fixed by the reference)
#define HID   1024
#define EMB   256
#define BATCH 64
#define GDIM  4096          // 4*HID
#define TGTV  32000

typedef __attribute__((ext_vector_type(16))) __bf16        v16bf;
typedef __attribute__((ext_vector_type(16))) unsigned short v16u;
typedef __attribute__((ext_vector_type(8)))  unsigned short v8u;
typedef __attribute__((ext_vector_type(8)))  float          v8f;

union BfFrag { v16u u; v16bf b; v8u h[2]; };

__device__ __forceinline__ unsigned short f2bfu(float f) {
    unsigned int u = __builtin_bit_cast(unsigned int, f);
    unsigned int r = (u + 0x7FFFu + ((u >> 16) & 1u)) >> 16;
    return (unsigned short)r;
}

__device__ __forceinline__ float sigm(float x) { return 1.0f / (1.0f + __expf(-x)); }

// ---------------------------------------------------------------------------
// bf16 GEMM, f32 accumulation:  C[m,n] = sum_k A[m,k]*B[n,k] + bias[n]
// A: bf16 [M,K] row-major, B: bf16 [N,K] row-major  (C = A @ B^T + bias)
// grid = (N/256, M/64), block = 256 threads (8 waves, 2(M) x 4(N)).
// Each wave: 2 M-tiles x 4 N-tiles (8 v8f accumulators); B fragments are
// loaded once from LDS and reused for both M-tiles (1 ds_load_b128 / WMMA).
// K-chunk = 64 -> 16 WMMAs per LDS-fill barrier pair.
// ---------------------------------------------------------------------------
__global__ __launch_bounds__(256)
void gemm_bf16_bias(const unsigned short* __restrict__ A,
                    const unsigned short* __restrict__ Bm,
                    const float* __restrict__ bias,
                    float* __restrict__ C, int N, int K)
{
    __shared__ unsigned short sB[256 * 72];   // 64 bf16 per row + 8 pad (16B-aligned stride)
    const int tid  = threadIdx.x;
    const int lane = tid & 31;
    const int wave = tid >> 5;
    const int wm   = wave >> 2;               // 0..1  : M pair select
    const int wn   = wave & 3;                // 0..3  : N quarter select
    const int ksel = lane >> 4;               // K-half select per 16-bit A/B layout
    const int nblock = blockIdx.x * 256;
    const int mrow0  = blockIdx.y * 64 + (wm * 2) * 16 + (lane & 15);

    v8f acc[2][4] = {};
    for (int k0 = 0; k0 < K; k0 += 64) {
        __syncthreads();
        // cooperative fill: B chunk [256 rows x 64 bf16] -> 2048 16B chunks / 256 thr
#pragma unroll
        for (int i = 0; i < 8; ++i) {
            int c    = tid + 256 * i;
            int row  = c >> 3;
            int part = c & 7;
            *(v8u*)(sB + row * 72 + part * 8) =
                *(const v8u*)(Bm + (size_t)(nblock + row) * K + k0 + part * 8);
        }
        __syncthreads();
#pragma unroll
        for (int ks = 0; ks < 2; ++ks) {
            // A fragments for both M-tiles:
            // lanes 0-15 hold K {0..7,16..23}; lanes 16-31 hold {8..15,24..31}
            BfFrag a0, a1;
            const unsigned short* pa =
                A + (size_t)mrow0 * K + k0 + ks * 32 + 8 * ksel;
            a0.h[0] = *(const v8u*)(pa);
            a0.h[1] = *(const v8u*)(pa + 16);
            const unsigned short* pa1 = pa + (size_t)16 * K;   // next 16 rows
            a1.h[0] = *(const v8u*)(pa1);
            a1.h[1] = *(const v8u*)(pa1 + 16);
#pragma unroll
            for (int t = 0; t < 4; ++t) {
                int nl = (wn * 4 + t) * 16 + (lane & 15);
                BfFrag bf;
                const unsigned short* pb = sB + nl * 72 + ks * 32 + ksel * 8;
                bf.h[0] = *(const v8u*)(pb);
                bf.h[1] = *(const v8u*)(pb + 16);
                acc[0][t] = __builtin_amdgcn_wmma_f32_16x16x32_bf16(
                    false, a0.b, false, bf.b, (short)0, acc[0][t], false, false);
                acc[1][t] = __builtin_amdgcn_wmma_f32_16x16x32_bf16(
                    false, a1.b, false, bf.b, (short)0, acc[1][t], false, false);
            }
        }
    }
    // Epilogue: D layout -> m = r + 8*(lane>=16), n = lane&15
#pragma unroll
    for (int mi = 0; mi < 2; ++mi) {
        const int mbase = blockIdx.y * 64 + (wm * 2 + mi) * 16 + 8 * ksel;
#pragma unroll
        for (int t = 0; t < 4; ++t) {
            int n = nblock + (wn * 4 + t) * 16 + (lane & 15);
            float bv = bias[n];
#pragma unroll
            for (int r = 0; r < 8; ++r)
                C[(size_t)(mbase + r) * N + n] = acc[mi][t][r] + bv;
        }
    }
}

// ---------------------------------------------------------------------------
// Fused LSTM step: gates = xg[t] + h_prev @ Whh_r^T (WMMA), then cell update.
// Gate columns are remapped: column (4*j+g) = gate g of hidden unit j, so one
// block (64 gate columns = 16 units) owns the full nonlinearity locally.
// grid = 64 blocks (gate-column slices), block = 128 threads (4 waves).
// ---------------------------------------------------------------------------
__global__ __launch_bounds__(128)
void lstm_step(const unsigned short* __restrict__ h_in,  // [64,1024] bf16
               const unsigned short* __restrict__ Whh,   // [4096,1024] bf16 remapped
               const float* __restrict__ xg_t,           // [64,4096] remapped (+bias)
               float* __restrict__ c,                    // [64,1024] f32 state
               unsigned short* __restrict__ h_out)       // [64,1024] bf16
{
    __shared__ unsigned short sB[64 * 40];
    __shared__ float gbuf[64 * 68];
    const int tid  = threadIdx.x;
    const int lane = tid & 31;
    const int wave = tid >> 5;                // = M tile (16 batch rows)
    const int ksel = lane >> 4;
    const int nb   = blockIdx.x * 64;         // gate-column base
    const int mrow = wave * 16 + (lane & 15);

    v8f acc[4] = {};
    for (int k0 = 0; k0 < HID; k0 += 32) {
        __syncthreads();
#pragma unroll
        for (int i = 0; i < 2; ++i) {         // 256 16B chunks / 128 thr
            int cc   = tid + 128 * i;
            int row  = cc >> 2;
            int part = cc & 3;
            *(v8u*)(sB + row * 40 + part * 8) =
                *(const v8u*)(Whh + (size_t)(nb + row) * HID + k0 + part * 8);
        }
        __syncthreads();
        BfFrag a;
        const unsigned short* pa = h_in + (size_t)mrow * HID + k0 + 8 * ksel;
        a.h[0] = *(const v8u*)(pa);
        a.h[1] = *(const v8u*)(pa + 16);
#pragma unroll
        for (int t = 0; t < 4; ++t) {
            int nl = t * 16 + (lane & 15);
            BfFrag bf;
            const unsigned short* pb = sB + nl * 40 + ksel * 8;
            bf.h[0] = *(const v8u*)(pb);
            bf.h[1] = *(const v8u*)(pb + 16);
            acc[t] = __builtin_amdgcn_wmma_f32_16x16x32_bf16(
                false, a.b, false, bf.b, (short)0, acc[t], false, false);
        }
    }
    __syncthreads();
    const int mbase = wave * 16 + 8 * ksel;
#pragma unroll
    for (int t = 0; t < 4; ++t) {
        int nl = t * 16 + (lane & 15);
#pragma unroll
        for (int r = 0; r < 8; ++r)
            gbuf[(mbase + r) * 68 + nl] = acc[t][r];
    }
    __syncthreads();
    // 64 batch x 16 units = 1024 cells / 128 threads
#pragma unroll
    for (int i = 0; i < 8; ++i) {
        int id = tid + 128 * i;
        int u = id & 15, b = id >> 4;
        int col = 4 * u;
        float ig = gbuf[b * 68 + col + 0] + xg_t[(size_t)b * GDIM + nb + col + 0];
        float fg = gbuf[b * 68 + col + 1] + xg_t[(size_t)b * GDIM + nb + col + 1];
        float gg = gbuf[b * 68 + col + 2] + xg_t[(size_t)b * GDIM + nb + col + 2];
        float og = gbuf[b * 68 + col + 3] + xg_t[(size_t)b * GDIM + nb + col + 3];
        int j  = (nb >> 2) + u;               // global hidden index
        int ci = b * HID + j;
        float cn = sigm(fg) * c[ci] + sigm(ig) * tanhf(gg);
        c[ci] = cn;
        float hn = sigm(og) * tanhf(cn);
        h_out[ci] = f2bfu(hn);
    }
}

// ---------------------------------------------------------------------------
// Repack / convert / gather / fill helpers
// ---------------------------------------------------------------------------
// W_r[(4j+g), k] = bf16(W[(g*H+j), k]); grid = (Kin/256, 4096), block 256
__global__ void pack_w(const float* __restrict__ W, unsigned short* __restrict__ out, int Kin)
{
    int ro = blockIdx.y;
    int k  = blockIdx.x * 256 + threadIdx.x;
    int j = ro >> 2, g = ro & 3;
    out[(size_t)ro * Kin + k] = f2bfu(W[(size_t)(g * HID + j) * Kin + k]);
}
__global__ void pack_bias(const float* __restrict__ bi, const float* __restrict__ bh,
                          float* __restrict__ out)
{
    int o = blockIdx.x * 256 + threadIdx.x;   // 4096 total
    int j = o >> 2, g = o & 3, s = g * HID + j;
    out[o] = bi[s] + bh[s];
}
__global__ void conv_bf16(const float* __restrict__ in, unsigned short* __restrict__ out, int n)
{
    int i = blockIdx.x * 256 + threadIdx.x;
    if (i < n) out[i] = f2bfu(in[i]);
}
// one block per (t,b) row; 256 threads = EMB columns
__global__ void gather_emb(const int* __restrict__ ids, const float* __restrict__ emb,
                           unsigned short* __restrict__ out)
{
    int row = blockIdx.x;
    int e   = threadIdx.x;
    out[(size_t)row * EMB + e] = f2bfu(emb[(size_t)ids[row] * EMB + e]);
}
__global__ void fill_f32(float* p, int n)
{ int i = blockIdx.x * 256 + threadIdx.x; if (i < n) p[i] = 0.0f; }
__global__ void fill_u16(unsigned short* p, int n)
{ int i = blockIdx.x * 256 + threadIdx.x; if (i < n) p[i] = 0; }

// ---------------------------------------------------------------------------
extern "C" void kernel_launch(void* const* d_in, const int* in_sizes, int n_in,
                              void* d_out, int out_size, void* d_ws, size_t ws_size,
                              hipStream_t stream)
{
    (void)in_sizes; (void)n_in; (void)out_size; (void)ws_size;
    const int*   src     = (const int*)d_in[0];
    const int*   tgt     = (const int*)d_in[1];
    const float* enc_emb = (const float*)d_in[2];
    const float* dec_emb = (const float*)d_in[3];
    const float* fc_W    = (const float*)d_in[4];
    const float* fc_b    = (const float*)d_in[5];
    const float* Wih[4]  = { (const float*)d_in[6],  (const float*)d_in[10],
                             (const float*)d_in[14], (const float*)d_in[18] }; // e0,e1,d0,d1
    const float* Whh[4]  = { (const float*)d_in[7],  (const float*)d_in[11],
                             (const float*)d_in[15], (const float*)d_in[19] };
    const float* bih[4]  = { (const float*)d_in[8],  (const float*)d_in[12],
                             (const float*)d_in[16], (const float*)d_in[20] };
    const float* bhh[4]  = { (const float*)d_in[9],  (const float*)d_in[13],
                             (const float*)d_in[17], (const float*)d_in[21] };
    const int Kin[4] = { EMB, HID, EMB, HID };
    float* out = (float*)d_out;

    // --- workspace carve-up (256B aligned) ---
    char* ws = (char*)d_ws;
    size_t off = 0;
    auto alloc = [&](size_t bytes) -> char* {
        char* p = ws + off;
        off = (off + bytes + 255) & ~(size_t)255;
        return p;
    };
    unsigned short* xe  = (unsigned short*)alloc((size_t)48 * 64 * EMB * 2);
    unsigned short* xd  = (unsigned short*)alloc((size_t)47 * 64 * EMB * 2);
    float*          xg  = (float*)alloc((size_t)3072 * GDIM * 4);
    unsigned short* ye0 = (unsigned short*)alloc((size_t)48 * 64 * HID * 2);
    unsigned short* ye1 = (unsigned short*)alloc((size_t)48 * 64 * HID * 2);
    unsigned short* yd0 = (unsigned short*)alloc((size_t)47 * 64 * HID * 2);
    unsigned short* yd1 = (unsigned short*)alloc((size_t)47 * 64 * HID * 2);
    float*          c0  = (float*)alloc((size_t)64 * HID * 4);
    float*          c1  = (float*)alloc((size_t)64 * HID * 4);
    unsigned short* h0z = (unsigned short*)alloc((size_t)64 * HID * 2);
    unsigned short* Wih_r[4];
    unsigned short* Whh_r[4];
    float*          br[4];
    for (int l = 0; l < 4; ++l) {
        Wih_r[l] = (unsigned short*)alloc((size_t)GDIM * Kin[l] * 2);
        Whh_r[l] = (unsigned short*)alloc((size_t)GDIM * HID * 2);
        br[l]    = (float*)alloc((size_t)GDIM * 4);
    }
    unsigned short* fcw = (unsigned short*)alloc((size_t)TGTV * HID * 2);

    // --- init state & output row 0 ---
    fill_f32<<<(64 * TGTV + 255) / 256, 256, 0, stream>>>(out, 64 * TGTV);
    fill_f32<<<(64 * HID + 255) / 256, 256, 0, stream>>>(c0, 64 * HID);
    fill_f32<<<(64 * HID + 255) / 256, 256, 0, stream>>>(c1, 64 * HID);
    fill_u16<<<(64 * HID + 255) / 256, 256, 0, stream>>>(h0z, 64 * HID);

    // --- weight repack (gate remap + bf16) ---
    for (int l = 0; l < 4; ++l) {
        pack_w<<<dim3(Kin[l] / 256, GDIM), 256, 0, stream>>>(Wih[l], Wih_r[l], Kin[l]);
        pack_w<<<dim3(HID / 256, GDIM), 256, 0, stream>>>(Whh[l], Whh_r[l], HID);
        pack_bias<<<GDIM / 256, 256, 0, stream>>>(bih[l], bhh[l], br[l]);
    }
    conv_bf16<<<((int)((size_t)TGTV * HID) + 255) / 256, 256, 0, stream>>>(
        fc_W, fcw, TGTV * HID);

    // --- embeddings ---
    gather_emb<<<48 * 64, 256, 0, stream>>>(src, enc_emb, xe);   // xe
    gather_emb<<<47 * 64, 256, 0, stream>>>(tgt, dec_emb, xd);   // target[:-1]

    // --- one LSTM layer: batched xg GEMM, then sequential fused steps ---
    auto run_layer = [&](const unsigned short* x, int T, int l,
                         const unsigned short* h_init, float* cbuf, unsigned short* y) {
        dim3 g(GDIM / 256, T);                 // M = T*64 -> T blocks of 64 rows
        gemm_bf16_bias<<<g, 256, 0, stream>>>(x, Wih_r[l], br[l], xg, GDIM, Kin[l]);
        for (int t = 0; t < T; ++t) {
            const unsigned short* hin =
                (t == 0) ? h_init : (y + (size_t)(t - 1) * 64 * HID);
            lstm_step<<<64, 128, 0, stream>>>(hin, Whh_r[l],
                                              xg + (size_t)t * 64 * GDIM,
                                              cbuf, y + (size_t)t * 64 * HID);
        }
    };

    run_layer(xe,  48, 0, h0z,                          c0, ye0);  // enc L0
    run_layer(ye0, 48, 1, h0z,                          c1, ye1);  // enc L1
    run_layer(xd,  47, 2, ye0 + (size_t)47 * 64 * HID,  c0, yd0);  // dec L0 (carries enc L0 h,c)
    run_layer(yd0, 47, 3, ye1 + (size_t)47 * 64 * HID,  c1, yd1);  // dec L1 (carries enc L1 h,c)

    // --- final projection: [3008,1024] x [32000,1024]^T + fc_b -> out rows 64.. ---
    dim3 gfc(TGTV / 256, 47);
    gemm_bf16_bias<<<gfc, 256, 0, stream>>>(yd1, fcw, fc_b,
                                            out + (size_t)64 * TGTV, TGTV, HID);
}
